// IpaScore_84731114816262
// MI455X (gfx1250) — matrix operand order repaired and checked by reading, not compiled, not run on hardware
//
#include <hip/hip_runtime.h>
#include <math.h>

typedef __attribute__((ext_vector_type(16))) _Float16 v16h;
typedef __attribute__((ext_vector_type(8)))  _Float16 v8h;
typedef __attribute__((ext_vector_type(4)))  _Float16 v4h;
typedef __attribute__((ext_vector_type(8)))  float    v8f;
typedef __attribute__((ext_vector_type(4)))  float    v4f;

// ---------------------------------------------------------------------------
// Generic WMMA GEMM:  Y[m][n] = alpha * sum_k X[m][k] * B(k,n) (+bias[n]) (+Y)
//   flags bit0: B(k,n) = W[n*ldW + k]   (weights stored [N][K], i.e. X @ W^T)
//         else: B(k,n) = W[k*ldW + n]
//   flags bit1: relu    flags bit2: accumulate into Y
// Block: 128 threads = 4 wave32s; block tile 64x64; K tile 32; LDS double-buffer.
// ---------------------------------------------------------------------------
__global__ __launch_bounds__(128) void k_gemm(
    const float* __restrict__ X, int ldX,
    const float* __restrict__ W, int ldW,
    const float* __restrict__ bias,
    float* __restrict__ Y, int ldY,
    int M, int N, int K, float alpha, int flags)
{
  const bool transB = flags & 1;
  const bool dorelu = flags & 2;
  const bool accum  = flags & 4;
  __shared__ _Float16 As[2][64][40];   // [m][k], 80B rows (16B aligned)
  __shared__ _Float16 Bs[2][64][40];   // [n][k]
  const int tid  = threadIdx.x;
  const int bm   = blockIdx.y * 64;
  const int bn   = blockIdx.x * 64;
  const int wv   = tid >> 5;           // wave32
  const int lane = tid & 31;
  const int wrow = wv >> 1, wcol = wv & 1;
  const int g    = lane >> 4, lm = lane & 15;
  const bool fullM = (bm + 64 <= M);
  const bool fullN = (bn + 64 <= N);

  auto loadA = [&](int buf, int k0) {
    if (fullM && (k0 + 32 <= K)) {
      if (k0 + 64 < K && tid < 64)
        __builtin_prefetch(X + (size_t)(bm + tid) * ldX + (k0 + 64), 0, 3);
#pragma unroll
      for (int it = 0; it < 4; ++it) {
        int e = (tid + it * 128) * 4;        // 0..2044, step 4
        int r = e >> 5, c = e & 31;
        v4f vv = *(const v4f*)(X + (size_t)(bm + r) * ldX + (k0 + c));
        v4h hh;
        hh[0] = (_Float16)vv[0]; hh[1] = (_Float16)vv[1];
        hh[2] = (_Float16)vv[2]; hh[3] = (_Float16)vv[3];
        *(v4h*)&As[buf][r][c] = hh;
      }
    } else {
#pragma unroll
      for (int it = 0; it < 16; ++it) {
        int idx = tid + it * 128;
        int r = idx >> 5, c = idx & 31;
        int gr = bm + r, gc = k0 + c;
        float v = 0.f;
        if (gr < M && gc < K) v = X[(size_t)gr * ldX + gc];
        As[buf][r][c] = (_Float16)v;
      }
    }
  };

  auto loadB = [&](int buf, int k0) {
    if (transB) {
      if (fullN && (k0 + 32 <= K)) {
        if (k0 + 64 < K && tid < 64)
          __builtin_prefetch(W + (size_t)(bn + tid) * ldW + (k0 + 64), 0, 3);
#pragma unroll
        for (int it = 0; it < 4; ++it) {
          int e = (tid + it * 128) * 4;
          int r = e >> 5, c = e & 31;
          v4f vv = *(const v4f*)(W + (size_t)(bn + r) * ldW + (k0 + c));
          v4h hh;
          hh[0] = (_Float16)vv[0]; hh[1] = (_Float16)vv[1];
          hh[2] = (_Float16)vv[2]; hh[3] = (_Float16)vv[3];
          *(v4h*)&Bs[buf][r][c] = hh;
        }
      } else {
#pragma unroll
        for (int it = 0; it < 16; ++it) {
          int idx = tid + it * 128;
          int n = idx >> 5, c = idx & 31;
          int gn = bn + n, gc = k0 + c;
          float v = 0.f;
          if (gn < N && gc < K) v = W[(size_t)gn * ldW + gc];
          Bs[buf][n][c] = (_Float16)v;
        }
      }
    } else {
      if (fullN && (k0 + 32 <= K)) {
#pragma unroll
        for (int it = 0; it < 4; ++it) {
          int e = (tid + it * 128) * 4;       // (c, n-quad) tiling: c = e>>6, n0 = e&63
          int c = e >> 6, n0 = e & 63;
          v4f vv = *(const v4f*)(W + (size_t)(k0 + c) * ldW + (bn + n0));
          Bs[buf][n0 + 0][c] = (_Float16)vv[0];
          Bs[buf][n0 + 1][c] = (_Float16)vv[1];
          Bs[buf][n0 + 2][c] = (_Float16)vv[2];
          Bs[buf][n0 + 3][c] = (_Float16)vv[3];
        }
      } else {
#pragma unroll
        for (int it = 0; it < 16; ++it) {
          int idx = tid + it * 128;
          int n = idx >> 5, c = idx & 31;
          int gn = bn + n, gc = k0 + c;
          float v = 0.f;
          if (gn < N && gc < K) v = W[(size_t)gc * ldW + gn];
          Bs[buf][n][c] = (_Float16)v;
        }
      }
    }
  };

  v8f acc[2][2];
#pragma unroll
  for (int a = 0; a < 2; ++a)
#pragma unroll
    for (int b = 0; b < 2; ++b) acc[a][b] = v8f{0.f,0.f,0.f,0.f,0.f,0.f,0.f,0.f};

  loadA(0, 0); loadB(0, 0);
  const int nk = (K + 31) >> 5;
  for (int ki = 0; ki < nk; ++ki) {
    __syncthreads();
    const int cur = ki & 1;
    if (ki + 1 < nk) { loadA(cur ^ 1, (ki + 1) * 32); loadB(cur ^ 1, (ki + 1) * 32); }

    v16h af[2], bf[2];
#pragma unroll
    for (int mi = 0; mi < 2; ++mi) {
      const _Float16* ap = &As[cur][wrow * 32 + mi * 16 + lm][8 * g];
      v16h a;
#pragma unroll
      for (int h = 0; h < 8; ++h) { a[h] = ap[h]; a[8 + h] = ap[16 + h]; }
      af[mi] = a;
    }
#pragma unroll
    for (int ni = 0; ni < 2; ++ni) {
      const _Float16* bp = &Bs[cur][wcol * 32 + ni * 16 + lm][8 * g];
      v16h bb;
#pragma unroll
      for (int h = 0; h < 8; ++h) { bb[h] = bp[h]; bb[8 + h] = bp[16 + h]; }
      bf[ni] = bb;
    }
#pragma unroll
    for (int mi = 0; mi < 2; ++mi)
#pragma unroll
      for (int ni = 0; ni < 2; ++ni)
        acc[mi][ni] = __builtin_amdgcn_wmma_f32_16x16x32_f16(
            false, af[mi], false, bf[ni], (short)0, acc[mi][ni], false, false);
  }

#pragma unroll
  for (int mi = 0; mi < 2; ++mi)
#pragma unroll
    for (int ni = 0; ni < 2; ++ni) {
      int col = bn + wcol * 32 + ni * 16 + lm;
      if (col >= N) continue;
      float bv = bias ? bias[col] : 0.f;
#pragma unroll
      for (int r = 0; r < 8; ++r) {
        int row = bm + wrow * 32 + mi * 16 + 8 * g + r;
        if (row >= M) continue;
        float v = alpha * acc[mi][ni][r] + bv;
        if (accum) v += Y[(size_t)row * ldY + col];
        if (dorelu) v = fmaxf(v, 0.f);
        Y[(size_t)row * ldY + col] = v;
      }
    }
}

// ---------------------------------------------------------------------------
// Async (ASYNCcnt-tracked) bulk copy through LDS: one 16B packet per lane.
// n must be an exact multiple of 256 lanes * 16B per block (true for z: 33.5MB).
// ---------------------------------------------------------------------------
__global__ void k_copy_async(float* __restrict__ dst, const float* __restrict__ src)
{
  __shared__ __align__(16) char stg[256 * 16];
  unsigned lds = (unsigned)(uintptr_t)(&stg[threadIdx.x * 16]);
  size_t byteoff = ((size_t)blockIdx.x * 256 + threadIdx.x) * 16;
  unsigned long long ga = (unsigned long long)(uintptr_t)src + byteoff;
  unsigned long long gd = (unsigned long long)(uintptr_t)dst + byteoff;
  asm volatile("global_load_async_to_lds_b128 %0, %1, off"
               :: "v"(lds), "v"(ga) : "memory");
  asm volatile("s_wait_asynccnt 0" ::: "memory");
  asm volatile("global_store_async_from_lds_b128 %0, %1, off"
               :: "v"(gd), "v"(lds) : "memory");
  asm volatile("s_wait_asynccnt 0" ::: "memory");
}

// ---------------------------------------------------------------------------
// Small VALU helper kernels
// ---------------------------------------------------------------------------
__device__ __forceinline__ void quat2rot(float w, float x, float y, float z, float* R)
{
  R[0] = 1.f - 2.f * (y * y + z * z); R[1] = 2.f * (x * y - w * z); R[2] = 2.f * (x * z + w * y);
  R[3] = 2.f * (x * y + w * z); R[4] = 1.f - 2.f * (x * x + z * z); R[5] = 2.f * (y * z - w * x);
  R[6] = 2.f * (x * z - w * y); R[7] = 2.f * (y * z + w * x); R[8] = 1.f - 2.f * (x * x + y * y);
}

__global__ void k_init_rigid(const float* __restrict__ rig, const float* __restrict__ mk,
                             const float* __restrict__ fx, float* R, float* t, float* dm)
{
  int i = threadIdx.x;
  const float* q = rig + i * 7;
  float w = q[0], x = q[1], y = q[2], z = q[3];
  float inv = rsqrtf(w * w + x * x + y * y + z * z);
  w *= inv; x *= inv; y *= inv; z *= inv;
  quat2rot(w, x, y, z, R + i * 9);
  t[i * 3 + 0] = q[4] * 0.1f; t[i * 3 + 1] = q[5] * 0.1f; t[i * 3 + 2] = q[6] * 0.1f;
  dm[i] = (1.f - fx[i]) * mk[i];
}

__global__ void k_copy(float* __restrict__ d, const float* __restrict__ s, size_t n)
{ size_t i = (size_t)blockIdx.x * 256 + threadIdx.x; if (i < n) d[i] = s[i]; }

__global__ void k_add(float* __restrict__ d, const float* __restrict__ s, size_t n)
{ size_t i = (size_t)blockIdx.x * 256 + threadIdx.x; if (i < n) d[i] += s[i]; }

__global__ void k_mul_rows(float* __restrict__ d, const float* __restrict__ s,
                           const float* __restrict__ rs, int rows, int cols)
{
  size_t i = (size_t)blockIdx.x * 256 + threadIdx.x;
  if (i < (size_t)rows * cols) { int r = (int)(i / cols); d[i] = s[i] * rs[r]; }
}

__global__ void k_copy_cols(float* __restrict__ d, int ldD, const float* __restrict__ s,
                            int ldS, int rows, int cols)
{
  size_t i = (size_t)blockIdx.x * 256 + threadIdx.x;
  if (i < (size_t)rows * cols) {
    int r = (int)(i / cols), c = (int)(i % cols);
    d[(size_t)r * ldD + c] = s[(size_t)r * ldS + c];
  }
}

// in-place point transform: p <- R_i * p + t_i   (buf layout [i][P*3])
__global__ void k_point_transform(const float* __restrict__ R, const float* __restrict__ t,
                                  float* __restrict__ buf, int P)
{
  int idx = blockIdx.x * 256 + threadIdx.x;
  if (idx >= 256 * P) return;
  int i = idx / P;
  float* p = buf + (size_t)idx * 3;
  float x = p[0], y = p[1], z = p[2];
  const float* Ri = R + i * 9;
  p[0] = Ri[0] * x + Ri[1] * y + Ri[2] * z + t[i * 3 + 0];
  p[1] = Ri[3] * x + Ri[4] * y + Ri[5] * z + t[i * 3 + 1];
  p[2] = Ri[6] * x + Ri[7] * y + Ri[8] * z + t[i * 3 + 2];
}

// a[h][i][j] += coef * zb[(i*256+j)*8 + h]
__global__ void k_add_zbias(float* __restrict__ a, const float* __restrict__ zb, float coef)
{
  int idx = blockIdx.x * 256 + threadIdx.x;
  if (idx >= 8 * 65536) return;
  int h = idx >> 16, ij = idx & 65535;
  a[idx] += coef * zb[(size_t)ij * 8 + h];
}

// a[h][i][j] += -0.5 * softplus(hw[h]) * sqrt(1/108) * sum_p |qp[i,h,p]-kp[j,h,p]|^2
__global__ void k_d2(float* __restrict__ a, const float* __restrict__ qpt,
                     const float* __restrict__ kvpt, const float* __restrict__ hw)
{
  int h = blockIdx.x >> 8, i = blockIdx.x & 255, j = threadIdx.x;
  float w = hw[h];
  float sp = (w > 20.f) ? w : log1pf(expf(w));
  float hwv = sp * sqrtf(1.f / 108.f);
  const float* qp = qpt + (size_t)i * 192 + h * 24;
  const float* kp = kvpt + (size_t)j * 480 + h * 60;
  float s = 0.f;
#pragma unroll
  for (int p = 0; p < 8; ++p) {
    float dx = qp[p * 3 + 0] - kp[p * 3 + 0];
    float dy = qp[p * 3 + 1] - kp[p * 3 + 1];
    float dz = qp[p * 3 + 2] - kp[p * 3 + 2];
    s += dx * dx + dy * dy + dz * dz;
  }
  a[((size_t)h << 16) + i * 256 + j] += -0.5f * hwv * s;
}

// per-row softmax over 256 cols. mode0: += (m_i*m_j-1)*1e9 ; mode1: += (1-m_j)
__global__ void k_softmax(float* __restrict__ a, const float* __restrict__ mk, int mode)
{
  __shared__ float red[256];
  int row = blockIdx.x, j = threadIdx.x;
  int i = row & 255;
  float v = a[(size_t)row * 256 + j];
  if (mode == 0) v += (mk[i] * mk[j] - 1.f) * 1e9f;
  else           v += (1.f - mk[j]);
  red[j] = v; __syncthreads();
  for (int s = 128; s > 0; s >>= 1) { if (j < s) red[j] = fmaxf(red[j], red[j + s]); __syncthreads(); }
  float m = red[0]; __syncthreads();
  float e = expf(v - m);
  red[j] = e; __syncthreads();
  for (int s = 128; s > 0; s >>= 1) { if (j < s) red[j] += red[j + s]; __syncthreads(); }
  a[(size_t)row * 256 + j] = e / red[0];
}

// LayerNorm row kernel. dst = LN(A + pre*B) * post
// mode0: pre=1 post=1 ; mode1: pre=mk[row] ; mode2: post=mk[row] ; mode3: post=mk[i]*mk[j]
__global__ void k_ln(float* __restrict__ dst, const float* __restrict__ A,
                     const float* __restrict__ B, const float* __restrict__ gg,
                     const float* __restrict__ bb, const float* __restrict__ mk,
                     int C, int mode)
{
  __shared__ float red[256];
  int row = blockIdx.x, tid = threadIdx.x;
  float pre = 1.f, post = 1.f;
  if (mode == 1) pre = mk[row];
  else if (mode == 2) post = mk[row];
  else if (mode == 3) post = mk[row >> 8] * mk[row & 255];
  int cnt = (C + 255) / 256;
  float tv[2]; tv[0] = 0.f; tv[1] = 0.f;
  float s = 0.f;
  for (int it = 0; it < cnt; ++it) {
    int c = it * 256 + tid; float x = 0.f;
    if (c < C) { x = A[(size_t)row * C + c]; if (B) x += pre * B[(size_t)row * C + c]; }
    tv[it] = x; s += x;
  }
  red[tid] = s; __syncthreads();
  for (int st = 128; st > 0; st >>= 1) { if (tid < st) red[tid] += red[tid + st]; __syncthreads(); }
  float mean = red[0] / (float)C; __syncthreads();
  float s2 = 0.f;
  for (int it = 0; it < cnt; ++it) {
    int c = it * 256 + tid;
    float d = (c < C) ? (tv[it] - mean) : 0.f;
    s2 += d * d;
  }
  red[tid] = s2; __syncthreads();
  for (int st = 128; st > 0; st >>= 1) { if (tid < st) red[tid] += red[tid + st]; __syncthreads(); }
  float rstd = rsqrtf(red[0] / (float)C + 1e-5f); __syncthreads();
  for (int it = 0; it < cnt; ++it) {
    int c = it * 256 + tid;
    if (c < C) dst[(size_t)row * C + c] = ((tv[it] - mean) * rstd * gg[c] + bb[c]) * post;
  }
}

// opair[i][h][c] = sum_j a[h][i][j] * z[i][j][c] ; write into cat cols 2432+
__global__ void k_opair(const float* __restrict__ a, const float* __restrict__ z,
                        float* __restrict__ cat)
{
  __shared__ float sa[2048];
  int i = blockIdx.x, tid = threadIdx.x;
  for (int idx = tid; idx < 2048; idx += 256) {
    int h = idx >> 8, j = idx & 255;
    sa[idx] = a[((size_t)h << 16) + i * 256 + j];
  }
  __syncthreads();
  for (int q = 0; q < 4; ++q) {
    int idx = q * 256 + tid;
    int h = idx >> 7, c = idx & 127;
    float acc = 0.f;
    const float* zr = z + ((size_t)i * 256) * 128 + c;
    const float* ar = sa + h * 256;
    for (int j = 0; j < 256; ++j) acc += ar[j] * zr[(size_t)j * 128];
    cat[(size_t)i * 3456 + 2432 + h * 128 + c] = acc;
  }
}

// local = R_i^T (op - t_i); write opf (cols 2048..) and opn (cols 2336..)
__global__ void k_op_post(const float* __restrict__ op, const float* __restrict__ R,
                          const float* __restrict__ t, float* __restrict__ cat)
{
  int idx = blockIdx.x * 256 + threadIdx.x;
  if (idx >= 256 * 96) return;
  int i = idx / 96, q = idx % 96, h = q / 12, p = q % 12;
  const float* v = op + (size_t)i * 288 + h * 36 + p * 3;
  float x = v[0] - t[i * 3 + 0], y = v[1] - t[i * 3 + 1], z = v[2] - t[i * 3 + 2];
  const float* Ri = R + i * 9;
  float lx = Ri[0] * x + Ri[3] * y + Ri[6] * z;
  float ly = Ri[1] * x + Ri[4] * y + Ri[7] * z;
  float lz = Ri[2] * x + Ri[5] * y + Ri[8] * z;
  float* cr = cat + (size_t)i * 3456;
  cr[2048 + 0 * 96 + h * 12 + p] = lx;
  cr[2048 + 1 * 96 + h * 12 + p] = ly;
  cr[2048 + 2 * 96 + h * 12 + p] = lz;
  cr[2336 + h * 12 + p] = sqrtf(lx * lx + ly * ly + lz * lz + 1e-8f);
}

__global__ void k_build_e(float* __restrict__ e, const float* __restrict__ z,
                          const float* __restrict__ nb, int r0, int rows)
{
  size_t idx = (size_t)blockIdx.x * 256 + threadIdx.x;
  if (idx >= (size_t)rows * 384) return;
  int rr = (int)(idx / 384), c = (int)(idx % 384);
  int row = r0 + rr, i = row >> 8, j = row & 255;
  float v;
  if (c < 128)      v = z[(size_t)row * 128 + c];
  else if (c < 256) v = nb[(size_t)i * 128 + (c - 128)];
  else              v = nb[(size_t)j * 128 + (c - 256)];
  e[(size_t)rr * 384 + c] = v;
}

__global__ void k_rigid_update(float* __restrict__ R, float* __restrict__ t,
                               const float* __restrict__ upd, const float* __restrict__ dm)
{
  int i = threadIdx.x;
  float d = dm[i];
  float u0 = upd[i * 6 + 0] * d, u1 = upd[i * 6 + 1] * d, u2 = upd[i * 6 + 2] * d;
  float u3 = upd[i * 6 + 3] * d, u4 = upd[i * 6 + 4] * d, u5 = upd[i * 6 + 5] * d;
  float inv = rsqrtf(1.f + u0 * u0 + u1 * u1 + u2 * u2);
  float Q[9]; quat2rot(inv, u0 * inv, u1 * inv, u2 * inv, Q);
  float Rl[9];
#pragma unroll
  for (int k = 0; k < 9; ++k) Rl[k] = R[i * 9 + k];
  t[i * 3 + 0] += Rl[0] * u3 + Rl[1] * u4 + Rl[2] * u5;
  t[i * 3 + 1] += Rl[3] * u3 + Rl[4] * u4 + Rl[5] * u5;
  t[i * 3 + 2] += Rl[6] * u3 + Rl[7] * u4 + Rl[8] * u5;
#pragma unroll
  for (int r = 0; r < 3; ++r)
#pragma unroll
    for (int c = 0; c < 3; ++c)
      R[i * 9 + r * 3 + c] = Rl[r * 3 + 0] * Q[0 * 3 + c] + Rl[r * 3 + 1] * Q[1 * 3 + c] + Rl[r * 3 + 2] * Q[2 * 3 + c];
}

// out row = [latent(128), un(2), psi(2), t/SCALE(3), R(9)]  (144)
__global__ void k_finalize(const float* __restrict__ lat, const float* __restrict__ un,
                           const float* __restrict__ t, const float* __restrict__ R,
                           float* __restrict__ out)
{
  int i = blockIdx.x, c = threadIdx.x;
  float v;
  if (c < 128) v = lat[(size_t)i * 128 + c];
  else if (c < 130) v = un[i * 2 + (c - 128)];
  else if (c < 132) {
    float a = un[i * 2], b = un[i * 2 + 1];
    float n = rsqrtf(fmaxf(a * a + b * b, 1e-8f));
    v = (c == 130 ? a : b) * n;
  } else if (c < 135) v = t[i * 3 + (c - 132)] * 10.0f;
  else v = R[i * 9 + (c - 135)];
  out[(size_t)i * 144 + c] = v;
}

// ---------------------------------------------------------------------------
// Orchestration
// ---------------------------------------------------------------------------
extern "C" void kernel_launch(void* const* d_in, const int* in_sizes, int n_in,
                              void* d_out, int out_size, void* d_ws, size_t ws_size,
                              hipStream_t stream)
{
  (void)in_sizes; (void)n_in; (void)out_size; (void)ws_size;
  const float* init_node_embed = (const float*)d_in[0];
  const float* edge_embed      = (const float*)d_in[1];
  const float* res_mask        = (const float*)d_in[2];
  const float* fixed_mask      = (const float*)d_in[3];
  const float* rigids          = (const float*)d_in[4];
  const float* latent_features = (const float*)d_in[5];
  int pi = 6;
  auto P = [&]() { return (const float*)d_in[pi++]; };   // params in tree_flatten (sorted-key) order

  char* wsb = (char*)d_ws;
  size_t off = 0;
  auto alloc = [&](size_t nf) -> float* {
    float* p = (float*)(wsb + off);
    off += ((nf * 4 + 255) & ~(size_t)255);
    return p;
  };

  float* Rm    = alloc(256 * 9);
  float* tbuf  = alloc(256 * 3);
  float* dmask = alloc(256);
  float* node  = alloc(256 * 256);
  float* inode = alloc(256 * 256);
  float* zA    = alloc((size_t)256 * 256 * 128);
  float* zB    = alloc((size_t)256 * 256 * 128);
  float* qbuf  = alloc(256 * 2048);
  float* kvbuf = alloc(256 * 4096);
  float* qpt   = alloc(256 * 192);
  float* kvpt  = alloc(256 * 480);
  float* abuf  = alloc((size_t)8 * 256 * 256);
  float* zbias = alloc((size_t)65536 * 8);
  float* cat   = alloc((size_t)256 * 3456);
  float* opb   = alloc(256 * 288);
  float* xbuf  = alloc(256 * 320);
  float* qkv   = alloc(256 * 960);
  float* obuf  = alloc(256 * 320);
  float* tA    = alloc(256 * 1024);
  float* tB    = alloc(256 * 1024);
  float* nbuf  = alloc(256 * 128);
  float* ebuf  = alloc((size_t)4096 * 384);
  float* h1b   = alloc((size_t)4096 * 384);
  float* h2b   = alloc((size_t)4096 * 384);
  float* updb  = alloc(256 * 6);
  float* cat2  = alloc(256 * 384);
  float* latb  = alloc(256 * 128);
  float* sbuf  = alloc(256 * 256);
  float* unb   = alloc(256 * 2);

  auto gemm = [&](const float* X, int ldX, const float* W, int ldW, const float* bias,
                  float* Y, int ldY, int M, int N, int K, float alpha,
                  bool trB, bool relu, bool acc) {
    dim3 grid((N + 63) / 64, (M + 63) / 64);
    k_gemm<<<grid, 128, 0, stream>>>(X, ldX, W, ldW, bias, Y, ldY, M, N, K, alpha,
                                     (trB ? 1 : 0) | (relu ? 2 : 0) | (acc ? 4 : 0));
  };
  auto ew = [](size_t n) { return dim3((unsigned)((n + 255) / 256)); };

  k_init_rigid<<<1, 256, 0, stream>>>(rigids, res_mask, fixed_mask, Rm, tbuf, dmask);
  k_mul_rows<<<ew(65536), 256, 0, stream>>>(inode, init_node_embed, res_mask, 256, 256);
  k_copy<<<ew(65536), 256, 0, stream>>>(node, inode, 65536);
  // 33.5 MB z init copy through the CDNA5 async global<->LDS path
  // (8388608 floats = 2097152 x 16B packets = 8192 blocks x 256 lanes, exact)
  k_copy_async<<<dim3(8192), 256, 0, stream>>>(zA, edge_embed);
  float* z = zA; float* znew = zB;

  const float qk_scale = 1.0f / sqrtf(3.0f * 256.0f);
  const float zb_coef  = sqrtf(1.0f / 3.0f);
  const float tf_scale = 1.0f / sqrtf(80.0f);

  for (int b = 0; b < 4; ++b) {
    // ---- params for this block (alphabetical leaf order) ----
    const float *bb_b = P(), *bb_w = P();
    const float *ef_b = 0, *ef_w = 0, *ei_b = 0, *ei_w = 0, *eln_b = 0, *eln_g = 0,
                *et1_b = 0, *et1_w = 0, *et2_b = 0, *et2_w = 0;
    if (b < 3) {
      ef_b = P(); ef_w = P(); ei_b = P(); ei_w = P(); eln_b = P(); eln_g = P();
      et1_b = P(); et1_w = P(); et2_b = P(); et2_w = P();
    }
    const float *ib_b = P(), *ib_w = P();
    const float *ihw = P();
    const float *ikv_b = P(), *ikv_w = P();
    const float *ikvp_b = P(), *ikvp_w = P();
    const float *iout_b = P(), *iout_w = P();
    const float *iq_b = P(), *iq_w = P();
    const float *iqp_b = P(), *iqp_w = P();
    const float *iln_b = P(), *iln_g = P();
    const float *n1_b = P(), *n1_w = P();
    const float *n2_b = P(), *n2_w = P();
    const float *n3_b = P(), *n3_w = P();
    const float *nln_b = P(), *nln_g = P();
    const float *po_b = P(), *po_w = P();
    const float *sk_b = P(), *sk_w = P();
    const float* tfp[2][12];   // ff1_b,ff1_w,ff2_b,ff2_w,in_b,in_w,ln1_b,ln1_g,ln2_b,ln2_g,out_b,out_w
    for (int l = 0; l < 2; ++l) for (int k = 0; k < 12; ++k) tfp[l][k] = P();

    // ---- IPA ----
    gemm(node, 256, iq_w, 256, iq_b, qbuf, 2048, 256, 2048, 256, 1.f, true, false, false);
    gemm(node, 256, ikv_w, 256, ikv_b, kvbuf, 4096, 256, 4096, 256, 1.f, true, false, false);
    gemm(node, 256, iqp_w, 256, iqp_b, qpt, 192, 256, 192, 256, 1.f, true, false, false);
    gemm(node, 256, ikvp_w, 256, ikvp_b, kvpt, 480, 256, 480, 256, 1.f, true, false, false);
    k_point_transform<<<ew(256 * 64), 256, 0, stream>>>(Rm, tbuf, qpt, 64);
    k_point_transform<<<ew(256 * 160), 256, 0, stream>>>(Rm, tbuf, kvpt, 160);
    for (int h = 0; h < 8; ++h)
      gemm(qbuf + h * 256, 2048, kvbuf + h * 512, 4096, nullptr,
           abuf + (size_t)h * 65536, 256, 256, 256, 256, qk_scale, true, false, false);
    gemm(z, 128, ib_w, 128, ib_b, zbias, 8, 65536, 8, 128, 1.f, true, false, false);
    k_add_zbias<<<ew(524288), 256, 0, stream>>>(abuf, zbias, zb_coef);
    k_d2<<<dim3(8 * 256), 256, 0, stream>>>(abuf, qpt, kvpt, ihw);
    k_softmax<<<dim3(8 * 256), 256, 0, stream>>>(abuf, res_mask, 0);
    for (int h = 0; h < 8; ++h)
      gemm(abuf + (size_t)h * 65536, 256, kvbuf + h * 512 + 256, 4096, nullptr,
           cat + h * 256, 3456, 256, 256, 256, 1.f, false, false, false);
    for (int h = 0; h < 8; ++h)
      gemm(abuf + (size_t)h * 65536, 256, kvpt + h * 60 + 24, 480, nullptr,
           opb + h * 36, 288, 256, 36, 256, 1.f, false, false, false);
    k_op_post<<<ew(256 * 96), 256, 0, stream>>>(opb, Rm, tbuf, cat);
    k_opair<<<dim3(256), 256, 0, stream>>>(abuf, z, cat);
    gemm(cat, 3456, iout_w, 3456, iout_b, tA, 256, 256, 256, 3456, 1.f, true, false, false);
    k_ln<<<dim3(256), 256, 0, stream>>>(node, node, tA, iln_g, iln_b, res_mask, 256, 1);

    // ---- transformer stack ----
    k_copy_cols<<<ew(65536), 256, 0, stream>>>(xbuf, 320, node, 256, 256, 256);
    gemm(inode, 256, sk_w, 256, sk_b, xbuf + 256, 320, 256, 64, 256, 1.f, true, false, false);
    for (int l = 0; l < 2; ++l) {
      gemm(xbuf, 320, tfp[l][5], 320, tfp[l][4], qkv, 960, 256, 960, 320, 1.f, true, false, false);
      for (int th = 0; th < 4; ++th)
        gemm(qkv + th * 80, 960, qkv + 320 + th * 80, 960, nullptr,
             abuf + (size_t)th * 65536, 256, 256, 256, 80, tf_scale, true, false, false);
      k_softmax<<<dim3(4 * 256), 256, 0, stream>>>(abuf, res_mask, 1);
      for (int th = 0; th < 4; ++th)
        gemm(abuf + (size_t)th * 65536, 256, qkv + 640 + th * 80, 960, nullptr,
             obuf + th * 80, 320, 256, 80, 256, 1.f, false, false, false);
      gemm(obuf, 320, tfp[l][11], 320, tfp[l][10], tA, 320, 256, 320, 320, 1.f, true, false, false);
      k_ln<<<dim3(256), 256, 0, stream>>>(xbuf, xbuf, tA, tfp[l][7], tfp[l][6], nullptr, 320, 0);
      gemm(xbuf, 320, tfp[l][1], 320, tfp[l][0], tA, 320, 256, 320, 320, 1.f, true, true, false);
      gemm(tA, 320, tfp[l][3], 320, tfp[l][2], tB, 320, 256, 320, 320, 1.f, true, false, false);
      k_ln<<<dim3(256), 256, 0, stream>>>(xbuf, xbuf, tB, tfp[l][9], tfp[l][8], nullptr, 320, 0);
    }
    gemm(xbuf, 320, po_w, 320, po_b, node, 256, 256, 256, 320, 1.f, true, false, true);

    // ---- node transition ----
    gemm(node, 256, n1_w, 256, n1_b, tA, 256, 256, 256, 256, 1.f, true, true, false);
    gemm(tA, 256, n2_w, 256, n2_b, tB, 256, 256, 256, 256, 1.f, true, true, false);
    gemm(tB, 256, n3_w, 256, n3_b, tA, 256, 256, 256, 256, 1.f, true, false, false);
    k_ln<<<dim3(256), 256, 0, stream>>>(node, node, tA, nln_g, nln_b, res_mask, 256, 2);

    // ---- backbone update ----
    k_mul_rows<<<ew(65536), 256, 0, stream>>>(tA, node, dmask, 256, 256);
    gemm(tA, 256, bb_w, 256, bb_b, updb, 6, 256, 6, 256, 1.f, true, false, false);
    k_rigid_update<<<dim3(1), 256, 0, stream>>>(Rm, tbuf, updb, dmask);

    // ---- edge transition (chunked to bound workspace) ----
    if (b < 3) {
      gemm(node, 256, ei_w, 256, ei_b, nbuf, 128, 256, 128, 256, 1.f, true, false, false);
      for (int r0 = 0; r0 < 65536; r0 += 4096) {
        k_build_e<<<ew((size_t)4096 * 384), 256, 0, stream>>>(ebuf, z, nbuf, r0, 4096);
        gemm(ebuf, 384, et1_w, 384, et1_b, h1b, 384, 4096, 384, 384, 1.f, true, true, false);
        gemm(h1b, 384, et2_w, 384, et2_b, h2b, 384, 4096, 384, 384, 1.f, true, true, false);
        k_add<<<ew((size_t)4096 * 384), 256, 0, stream>>>(h2b, ebuf, (size_t)4096 * 384);
        gemm(h2b, 384, ef_w, 384, ef_b, znew + (size_t)r0 * 128, 128, 4096, 128, 384,
             1.f, true, false, false);
      }
      k_ln<<<dim3(65536), 256, 0, stream>>>(znew, znew, nullptr, eln_g, eln_b, res_mask, 128, 3);
      float* tz = z; z = znew; znew = tz;
    }
  }

  // ---- latent head ----
  const float *ll1_b = P(), *ll1_w = P(), *ll2_b = P(), *ll2_w = P(),
              *ll3_b = P(), *ll3_w = P(), *ll4_b = P(), *ll4_w = P(),
              *lln_b = P(), *lln_g = P();
  const float *pt1_b = P(), *pt1_w = P(), *pt2_b = P(), *pt2_w = P(),
              *plf_b = P(), *plf_w = P();

  k_copy_cols<<<ew(65536), 256, 0, stream>>>(cat2, 384, node, 256, 256, 256);
  k_copy_cols<<<ew(32768), 256, 0, stream>>>(cat2 + 256, 384, latent_features, 128, 256, 128);
  gemm(cat2, 384, ll1_w, 384, ll1_b, tA, 256, 256, 256, 384, 1.f, true, true, false);
  gemm(tA, 256, ll2_w, 256, ll2_b, tB, 256, 256, 256, 256, 1.f, true, true, false);
  gemm(tB, 256, ll3_w, 256, ll3_b, tA, 256, 256, 256, 256, 1.f, true, false, false);
  k_ln<<<dim3(256), 256, 0, stream>>>(tA, tA, nullptr, lln_g, lln_b, nullptr, 256, 0);
  k_copy<<<ew(32768), 256, 0, stream>>>(latb, latent_features, 32768);
  gemm(tA, 256, ll4_w, 256, ll4_b, latb, 128, 256, 128, 256, 1.f, true, false, true);

  // ---- torsion head ----
  gemm(node, 256, pt1_w, 256, pt1_b, tA, 256, 256, 256, 256, 1.f, true, true, false);
  k_copy<<<ew(65536), 256, 0, stream>>>(sbuf, node, 65536);
  gemm(tA, 256, pt2_w, 256, pt2_b, sbuf, 256, 256, 256, 256, 1.f, true, false, true);
  gemm(sbuf, 256, plf_w, 256, plf_b, unb, 2, 256, 2, 256, 1.f, true, false, false);

  k_finalize<<<dim3(256), 144, 0, stream>>>(latb, unb, tbuf, Rm, (float*)d_out);
}
